// SSMADense_29669634080851
// MI455X (gfx1250) — matrix operand order repaired and checked by reading, not compiled, not run on hardware
//
#include <hip/hip_runtime.h>

// ---------------- problem constants ----------------
#define Bb    2
#define Nn    1024
#define Ff    64
#define Kk    8
#define Hh    4
#define NC    32      // K*H attention channels
#define M1c   9
#define M2c   505
#define BINS  4545    // M1*M2
#define KD    9090    // 2*BINS (re,im interleaved)
#define KDP   9120    // padded to 32*285
#define KSTEPS 285
#define NODES 2048    // B*N
#define ECOLS 1024    // padded 2*M2 = 1010 -> 1024

// ---------------- WMMA plumbing (CDNA5 / gfx1250) ----------------
typedef __bf16        v16bf  __attribute__((ext_vector_type(16)));
typedef float         v8f    __attribute__((ext_vector_type(8)));
typedef unsigned int  u32x4  __attribute__((ext_vector_type(4)));

union BF16x16 {
  v16bf          v;
  unsigned short u[16];
  unsigned int   w[8];
  u32x4          q[2];
};

__device__ __forceinline__ unsigned short f2bf(float x) {
  union { float f; unsigned u; } v; v.f = x;
  unsigned r = v.u + 0x7FFFu + ((v.u >> 16) & 1u);   // RNE
  return (unsigned short)(r >> 16);
}

__device__ __forceinline__ v8f wmma_bf16(const BF16x16& a, const BF16x16& b, v8f c) {
  return __builtin_amdgcn_wmma_f32_16x16x32_bf16(false, a.v, false, b.v, (short)0, c, false, false);
}

// A-matrix 16x32 bf16 fragment (ISA 7.12.2): lane m=lane&15; lanes<16: K pairs
// {0,1..6,7 | 16..23}; lanes>=16: {8..15 | 24..31}. Pairs loaded as u32.
__device__ __forceinline__ void load_a_frag(BF16x16& a, const unsigned short* base,
                                            int ld, int col0, int lane) {
  const int m  = lane & 15;
  const int h8 = (lane & 16) ? 8 : 0;
  const unsigned short* row = base + m * ld + col0;
#pragma unroll
  for (int rr = 0; rr < 8; ++rr) {
    const int kk = (rr < 4) ? (h8 + 2 * rr) : (16 + h8 + 2 * (rr - 4));
    a.w[rr] = *(const unsigned int*)(row + kk);
  }
}

// B-matrix 32x16 bf16 fragment: lane n=lane&15 holds column n; K in same order as A.
__device__ __forceinline__ void load_b_frag(BF16x16& b, const unsigned short* base,
                                            int ld, int lane) {
  const int n  = lane & 15;
  const int h8 = (lane & 16) ? 8 : 0;
#pragma unroll
  for (int s = 0; s < 16; ++s) {
    const int kk = (s < 8) ? (h8 + s) : (16 + h8 + (s - 8));
    b.u[s] = base[kk * ld + n];
  }
}

// ---------------- small prep kernels ----------------

// xl = x@Wl+bl, xr = x@Wr+br  -> [B,N,32] f32 each
__global__ void k_lr(const float* __restrict__ x, const float* __restrict__ Wl,
                     const float* __restrict__ bl, const float* __restrict__ Wr,
                     const float* __restrict__ br, float* __restrict__ xl,
                     float* __restrict__ xr) {
  int gid = blockIdx.x * 256 + threadIdx.x;           // 131072
  int side = gid >> 16;
  int r = gid & 65535;
  int b = r >> 15, i = (r >> 5) & 1023, c = r & 31;
  const float* W  = side ? Wr : Wl;
  const float* bs = side ? br : bl;
  const float* xp = x + (size_t)((b << 10) + i) * Ff;
  float acc = bs[c];
#pragma unroll 8
  for (int f = 0; f < Ff; ++f) acc += xp[f] * W[f * NC + c];
  (side ? xr : xl)[(size_t)((b << 10) + i) * NC + c] = acc;
}

// Xh[b][g][j][kf] = bf16(x[b,j,g*16+kf])
__global__ void k_xh(const float* __restrict__ x, unsigned short* __restrict__ Xh) {
  int gid = blockIdx.x * 256 + threadIdx.x;           // 131072
  int b = gid >> 16, r = gid & 65535;
  int g = r >> 14, j = (r >> 4) & 1023, kf = r & 15;
  Xh[gid] = f2bf(x[(size_t)((b << 10) + j) * Ff + g * 16 + kf]);
}

// E[m2][2*f2] = cos(2pi f2 m2/505); E[m2][2*f2+1] = -sin(...)  (forward DFT)
__global__ void k_E(unsigned short* __restrict__ E) {
  int gid = blockIdx.x * 256 + threadIdx.x;           // 65536
  int m2 = gid >> 10, col = gid & 1023;
  float val = 0.f;
  if (col < 2 * M2c) {
    int f2 = col >> 1;
    int ph = (f2 * m2) % M2c;
    float ang = (float)ph * (6.28318530718f / (float)M2c);
    float sv, cv; __sincosf(ang, &sv, &cv);
    val = (col & 1) ? -sv : cv;
  }
  E[gid] = f2bf(val);
}

// C1[f][m1][f2] = sum_m2 Wm[m1*505+m2, f] * e^{+2pi i f2 m2/505}   (complex f32)
__global__ void k_c1(const float* __restrict__ Wm, float* __restrict__ C1) {
  int t = blockIdx.x * 256 + threadIdx.x;
  if (t >= Ff * M1c * M2c) return;
  int f  = t / (M1c * M2c);
  int r  = t - f * (M1c * M2c);
  int m1 = r / M2c, f2 = r - m1 * M2c;
  float ar = 0.f, ai = 0.f;
  for (int m2 = 0; m2 < M2c; ++m2) {
    int ph = (f2 * m2) % M2c;
    float ang = (float)ph * (6.28318530718f / (float)M2c);
    float sw, cw; __sincosf(ang, &sw, &cw);
    float w = Wm[(size_t)(m1 * M2c + m2) * Ff + f];
    ar += w * cw; ai += w * sw;
  }
  C1[(size_t)t * 2]     = ar;
  C1[(size_t)t * 2 + 1] = ai;
}

// Wt[2*(f1*505+f2)][f]   = Re(FFT2+(Wm))/4545
// Wt[2*(f1*505+f2)+1][f] = -Im(FFT2+(Wm))/4545   (minus folded for Real(Y*W))
__global__ void k_wt(const float* __restrict__ C1, unsigned short* __restrict__ Wt) {
  int t = blockIdx.x * 256 + threadIdx.x;             // 9120*64
  if (t >= KDP * Ff) return;
  int row = t >> 6, f = t & 63;
  float val = 0.f;
  if (row < KD) {
    int idx = row >> 1;
    int f1 = idx / M2c, f2 = idx - f1 * M2c;
    float wr = 0.f, wi = 0.f;
    for (int m1 = 0; m1 < M1c; ++m1) {
      int ph = (f1 * m1) % M1c;
      float ang = (float)ph * (6.28318530718f / (float)M1c);
      float sw, cw; __sincosf(ang, &sw, &cw);
      const float* c1p = C1 + (size_t)(((f * M1c) + m1) * M2c + f2) * 2;
      float cr = c1p[0], ci = c1p[1];
      wr += cr * cw - ci * sw;
      wi += cr * sw + ci * cw;
    }
    val = ((row & 1) ? -wi : wr) * (1.0f / (float)BINS);
  }
  Wt[(size_t)row * Ff + f] = f2bf(val);
}

// ---------------- fused attention + WMMA aggregation ----------------
// grid = B*32ch*64 i-tiles = 4096, block = 256 (8 waves).
// LDS: Xh tile 32KB | q f32 64KB | att bf16 32KB | xl column 4KB
__global__ void k_attn_agg(const float* __restrict__ xl, const float* __restrict__ xr,
                           const float* __restrict__ adj, const float* __restrict__ mask,
                           const unsigned short* __restrict__ Xh,
                           unsigned short* __restrict__ aggneg) {
  extern __shared__ char smem[];
  unsigned short* sh_xh  = (unsigned short*)smem;                     // 1024*16 bf16
  float*          sh_q   = (float*)(smem + 32768);                    // 16*1024 f32
  unsigned short* sh_att = (unsigned short*)(smem + 32768 + 65536);   // 16*1024 bf16
  float*          sh_xlc = (float*)(smem + 32768 + 65536 + 32768);    // 1024 f32

  int bx = blockIdx.x;
  int it = bx & 63, c = (bx >> 6) & 31, b = bx >> 11;
  int g = c & 3, n = c >> 2;
  int i0 = it << 4;
  int tid = threadIdx.x, lane = tid & 31;
  int r = tid >> 4, tr = tid & 15;

  // stage Xh[b][g] : 1024x16 bf16
  const unsigned short* gxh = Xh + ((size_t)(b * Hh + g) << 14);
  for (int e = tid; e < 2048; e += 256)
    ((u32x4*)sh_xh)[e] = ((const u32x4*)gxh)[e];

  float mi  = mask[(b << 10) + i0 + r];
  float xrv = xr[(size_t)((b << 10) + i0 + r) * NC + c];

  // pass 1: row max of leaky_relu(xl[j]+xr[i]) (masked)
  float mx = -3.0e38f;
  for (int j = tr; j < Nn; j += 16) {
    float xlv = xl[(size_t)((b << 10) + j) * NC + c];
    if (r == 0) sh_xlc[j] = xlv;
    float s = xlv + xrv;
    s = (s < 0.f) ? 0.01f * s : s;
    if (mi * mask[(b << 10) + j] == 0.f) s = -1e9f;
    mx = fmaxf(mx, s);
  }
#pragma unroll
  for (int m = 1; m < 16; m <<= 1) mx = fmaxf(mx, __shfl_xor(mx, m, 16));
  __syncthreads();   // sh_xlc + sh_xh visible

  // pass 2: e = exp(s-mx); q = e*adj
  float se = 0.f;
  const float* adjrow = adj + (size_t)((b << 10) + i0 + r) * Nn;
  for (int j = tr; j < Nn; j += 16) {
    float s = sh_xlc[j] + xrv;
    s = (s < 0.f) ? 0.01f * s : s;
    if (mi * mask[(b << 10) + j] == 0.f) s = -1e9f;
    float e = __expf(s - mx);
    se += e;
    sh_q[(r << 10) + j] = e * adjrow[j];
  }
#pragma unroll
  for (int m = 1; m < 16; m <<= 1) se += __shfl_xor(se, m, 16);
  float thr = 1e-6f * se;

  // pass 3: denom of clamped weights, then normalized bf16 attention
  float d = 0.f;
  for (int j = tr; j < Nn; j += 16) d += fmaxf(sh_q[(r << 10) + j], thr);
#pragma unroll
  for (int m = 1; m < 16; m <<= 1) d += __shfl_xor(d, m, 16);
  float invd = 1.f / d;
  for (int j = tr; j < Nn; j += 16)
    sh_att[(r << 10) + j] = f2bf(fmaxf(sh_q[(r << 10) + j], thr) * invd);
  __syncthreads();

  // WMMA: C[16 i x 16 kf] = sum over 32 j-chunks  att[16x32] x Xh[32x16]
  if (tid < 32) {
    v8f acc = {0.f, 0.f, 0.f, 0.f, 0.f, 0.f, 0.f, 0.f};
    for (int jc = 0; jc < 32; ++jc) {
      BF16x16 a, bm_;
      load_a_frag(a, sh_att, Nn, jc * 32, lane);
      load_b_frag(bm_, sh_xh + (jc * 32) * 16, 16, lane);
      acc = wmma_bf16(a, bm_, acc);
    }
    int Mb = (lane & 16) ? 8 : 0;
    int ncol = lane & 15;
#pragma unroll
    for (int rr = 0; rr < 8; ++rr) {
      int M = rr + Mb;
      // store NEGATED agg (x_aff row m1=0 is -agg)
      aggneg[(size_t)((b << 10) + i0 + M) * (Kk * Ff) + n * Ff + g * 16 + ncol] =
          f2bf(-acc[rr]);
    }
  }
}

// ---------------- per-node: S-DFT GEMM + spectral combine -> Y ----------------
// grid = 2048 nodes, block = 256 (8 waves). LDS: A 2KB + S 32KB
__global__ void k_node(const unsigned short* __restrict__ aggneg,
                       const unsigned short* __restrict__ E,
                       unsigned short* __restrict__ Y) {
  extern __shared__ char smem[];
  unsigned short* sA = (unsigned short*)smem;         // 16x64 bf16 (rows 8..15 zero)
  float*          sS = (float*)(smem + 2048);         // 8 x 1024 f32 (re,im interleaved)
  int node = blockIdx.x, tid = threadIdx.x, lane = tid & 31, w = tid >> 5;

  for (int e = tid; e < 1024; e += 256)
    sA[e] = (e < Kk * Ff) ? aggneg[(size_t)node * (Kk * Ff) + e] : (unsigned short)0;
  __syncthreads();

  // S[kch][col] = sum_m2 aggneg[kch][m2] * E[m2][col]; 64 N-tiles split over 8 waves
  for (int t = w; t < 64; t += 8) {
    v8f acc = {0.f, 0.f, 0.f, 0.f, 0.f, 0.f, 0.f, 0.f};
#pragma unroll
    for (int ks = 0; ks < 2; ++ks) {
      BF16x16 a, bm_;
      load_a_frag(a, sA, Ff, ks * 32, lane);
      load_b_frag(bm_, E + (size_t)(ks * 32) * ECOLS + 16 * t, ECOLS, lane);
      acc = wmma_bf16(a, bm_, acc);
    }
    int Mb = (lane & 16) ? 8 : 0;
    int ncol = lane & 15;
#pragma unroll
    for (int rr = 0; rr < 8; ++rr) {
      int m = rr + Mb;
      if (m < Kk) sS[(m << 10) + 16 * t + ncol] = acc[rr];
    }
  }
  __syncthreads();

  // X[k,f1,f2] = S[k][f2] + e^{-2pi i f1/9}; geometric-mean magnitude + summed phase
  for (int bin = tid; bin < BINS; bin += 256) {
    int f1 = bin / M2c, f2 = bin - f1 * M2c;
    float s1, c1; __sincosf((float)f1 * (6.28318530718f / (float)M1c), &s1, &c1);
    float w1r = c1, w1i = -s1;
    float lm = 0.f, an = 0.f;
#pragma unroll
    for (int k = 0; k < Kk; ++k) {
      float Xr = sS[(k << 10) + 2 * f2]     + w1r;
      float Xi = sS[(k << 10) + 2 * f2 + 1] + w1i;
      lm += __logf(sqrtf(Xr * Xr + Xi * Xi) + 1e-6f);
      an += atan2f(Xi, Xr);
    }
    float mag = __expf(lm * (1.0f / (float)Kk));
    float sy, cy; __sincosf(an, &sy, &cy);
    Y[(size_t)node * KDP + 2 * bin]     = f2bf(mag * cy);
    Y[(size_t)node * KDP + 2 * bin + 1] = f2bf(mag * sy);
  }
  for (int p2 = KD + tid; p2 < KDP; p2 += 256) Y[(size_t)node * KDP + p2] = 0;
}

// ---------------- final spectral projection GEMM ----------------
// out[2048 x 64] = Y[2048 x 9120] * Wt[9120 x 64]; grid = 128 M-tiles, block = 128
__global__ void k_out(const unsigned short* __restrict__ Y,
                      const unsigned short* __restrict__ Wt,
                      const float* __restrict__ bm, const float* __restrict__ mask,
                      float* __restrict__ out) {
  extern __shared__ char smem[];
  unsigned int* sY = (unsigned int*)smem;             // 16x32 bf16 A chunk
  int i0 = blockIdx.x << 4;
  int tid = threadIdx.x, lane = tid & 31, w = tid >> 5;   // w: N-tile 0..3
  v8f acc = {0.f, 0.f, 0.f, 0.f, 0.f, 0.f, 0.f, 0.f};
  for (int s = 0; s < KSTEPS; ++s) {
    for (int u = tid; u < 256; u += 128) {
      int row = u >> 4, cp = u & 15;
      sY[u] = *(const unsigned int*)(Y + (size_t)(i0 + row) * KDP + s * 32 + cp * 2);
    }
    if (s + 1 < KSTEPS)   // gfx1250 global_prefetch path for next A panel
      __builtin_prefetch(Y + (size_t)i0 * KDP + (s + 1) * 32, 0, 0);
    __syncthreads();
    BF16x16 a, bm_;
    load_a_frag(a, (const unsigned short*)sY, 32, 0, lane);
    load_b_frag(bm_, Wt + (size_t)(s * 32) * Ff + 16 * w, Ff, lane);
    acc = wmma_bf16(a, bm_, acc);
    __syncthreads();
  }
  int Mb = (lane & 16) ? 8 : 0;
  int ncol = lane & 15;
#pragma unroll
  for (int rr = 0; rr < 8; ++rr) {
    int node = i0 + rr + Mb;
    int f = 16 * w + ncol;
    out[(size_t)node * Ff + f] = (acc[rr] + bm[f]) * mask[node];
  }
}

// ---------------- host launch ----------------
extern "C" void kernel_launch(void* const* d_in, const int* in_sizes, int n_in,
                              void* d_out, int out_size, void* d_ws, size_t ws_size,
                              hipStream_t stream) {
  (void)in_sizes; (void)n_in; (void)out_size; (void)ws_size;
  const float* x    = (const float*)d_in[0];
  const float* adj  = (const float*)d_in[1];
  const float* mask = (const float*)d_in[2];
  const float* Wl   = (const float*)d_in[3];
  const float* bl   = (const float*)d_in[4];
  const float* Wr   = (const float*)d_in[5];
  const float* br   = (const float*)d_in[6];
  // d_in[7]=aff_w, d_in[8]=aff_b: fixed structure (-I block, delta at M2) folded in.
  const float* Wm   = (const float*)d_in[9];
  const float* bm   = (const float*)d_in[10];
  float* out = (float*)d_out;

  char* p = (char*)d_ws;
  float*          xl  = (float*)p;          p += (size_t)NODES * NC * 4;
  float*          xr  = (float*)p;          p += (size_t)NODES * NC * 4;
  unsigned short* Xh  = (unsigned short*)p; p += (size_t)Bb * Hh * Nn * 16 * 2;
  unsigned short* E   = (unsigned short*)p; p += (size_t)64 * ECOLS * 2;
  unsigned short* agn = (unsigned short*)p; p += (size_t)NODES * Kk * Ff * 2;
  float*          C1  = (float*)p;          p += (size_t)Ff * M1c * M2c * 2 * 4;
  unsigned short* Wt  = (unsigned short*)p; p += (size_t)KDP * Ff * 2;
  unsigned short* Y   = (unsigned short*)p; p += (size_t)NODES * KDP * 2;

  const size_t SMEM_ATTN = 32768 + 65536 + 32768 + 4096;  // 135168
  const size_t SMEM_NODE = 2048 + 32768;                  // 34816
  const size_t SMEM_OUT  = 1024;

  k_lr<<<512, 256, 0, stream>>>(x, Wl, bl, Wr, br, xl, xr);
  k_xh<<<512, 256, 0, stream>>>(x, Xh);
  k_E<<<256, 256, 0, stream>>>(E);
  k_c1<<<(Ff * M1c * M2c + 255) / 256, 256, 0, stream>>>(Wm, C1);
  k_wt<<<(KDP * Ff + 255) / 256, 256, 0, stream>>>(C1, Wt);
  k_attn_agg<<<4096, 256, SMEM_ATTN, stream>>>(xl, xr, adj, mask, Xh, agn);
  k_node<<<2048, 256, SMEM_NODE, stream>>>(agn, E, Y);
  k_out<<<128, 128, SMEM_OUT, stream>>>(Y, Wt, bm, mask, out);
}
